// GCNEncoder_26929444946547
// MI455X (gfx1250) — matrix operand order repaired and checked by reading, not compiled, not run on hardware
//
#include <hip/hip_runtime.h>

typedef __attribute__((ext_vector_type(16))) __bf16 v16bf;
typedef __attribute__((ext_vector_type(8)))  __bf16 v8bf;
typedef __attribute__((ext_vector_type(8)))  float  v8f;

#define WAVES_PER_BLOCK 8

// Assemble a 16-element bf16 WMMA operand fragment from two contiguous
// 16-byte chunks (per the CDNA5 A/B VGPR layouts).
__device__ __forceinline__ v16bf ld_frag(const __bf16* p, int o0, int o1)
{
    v8bf lo = *(const v8bf*)(p + o0);
    v8bf hi = *(const v8bf*)(p + o1);
    v16bf v;
#pragma unroll
    for (int i = 0; i < 8; ++i) { v[i] = lo[i]; v[i + 8] = hi[i]; }
    return v;
}

// Generic bf16 WMMA GEMM with 2x4 register blocking:
// each wave computes a 32x64 output block (8 independent 16x16 accumulators).
// D[b] = act( scale * A[b](MxK,row-major) * B + bias ), B given TRANSPOSED:
// BT[b] is (N x K) row-major so every lane's B fragment is contiguous.
// Requires M % 32 == 0, N % 64 == 0, K % 32 == 0 (true for all calls here).
// flags: 1 = add bias[col], 2 = ReLU, 4 = store bf16 (else f32)
__global__ void __launch_bounds__(256)
gcn_wmma_gemm(const __bf16* __restrict__ A, const __bf16* __restrict__ BT,
              const float* __restrict__ bias, void* __restrict__ Dst,
              int M, int Nn, int K,
              long sA, long sB, long sD, int batch,
              float scale, int flags)
{
    int wave  = blockIdx.x * WAVES_PER_BLOCK + (threadIdx.x >> 5);
    int bn    = Nn >> 6;                 // 64-wide column blocks
    int per   = (M >> 5) * bn;           // 32-tall row blocks
    int total = per * batch;
    if (wave >= total) return;
    int b  = wave / per;
    int t  = wave - b * per;
    int tm = t / bn;
    int tn = t - tm * bn;

    int lane = threadIdx.x & 31;
    int half = lane >> 4;
    int r    = lane & 15;

    const __bf16* A0 = A  + (long)b * sA + (long)(tm * 32 + r) * K;       // M tile 0
    const __bf16* A1 = A0 + (long)16 * K;                                 // M tile 1
    const __bf16* B0 = BT + (long)b * sB + (long)(tn * 64 + r) * K;      // N tile 0
    const long   bstep = (long)16 * K;                                    // next N tile

    // A 16x32 bf16 layout: lanes 0-15 hold K chunks [0,8)+[16,24),
    //                      lanes 16-31 hold [8,16)+[24,32)
    const int a0 = half * 8;
    const int a1 = 16 + half * 8;
    // B 32x16 bf16 layout: same K chunks as A ([0,8)+[16,24) / [8,16)+[24,32))
    const int b0 = half * 8;

    v8f acc[2][4];
#pragma unroll
    for (int i = 0; i < 2; ++i)
#pragma unroll
        for (int j = 0; j < 4; ++j)
            acc[i][j] = (v8f){0.f, 0.f, 0.f, 0.f, 0.f, 0.f, 0.f, 0.f};

    for (int k = 0; k < K; k += 32) {
        v16bf av[2], bv[4];
        av[0] = ld_frag(A0 + k, a0, a1);
        av[1] = ld_frag(A1 + k, a0, a1);
#pragma unroll
        for (int j = 0; j < 4; ++j)
            bv[j] = ld_frag(B0 + j * bstep + k, b0, b0 + 16);
#pragma unroll
        for (int i = 0; i < 2; ++i)
#pragma unroll
            for (int j = 0; j < 4; ++j)
                acc[i][j] = __builtin_amdgcn_wmma_f32_16x16x32_bf16(
                    false, av[i], false, bv[j], (short)0, acc[i][j], false, false);
    }

    // C/D layout: lane half selects rows [0,8) vs [8,16); VGPR v = row in half
#pragma unroll
    for (int j = 0; j < 4; ++j) {
        int col = tn * 64 + j * 16 + r;
        float bb = (flags & 1) ? bias[col] : 0.f;
#pragma unroll
        for (int i = 0; i < 2; ++i) {
            int row0 = tm * 32 + i * 16 + half * 8;
            if (flags & 4) {
                __bf16* Dp = (__bf16*)Dst + (long)b * sD;
#pragma unroll
                for (int v = 0; v < 8; ++v) {
                    float x = acc[i][j][v] * scale + bb;
                    if (flags & 2) x = fmaxf(x, 0.f);
                    Dp[(long)(row0 + v) * Nn + col] = (__bf16)x;
                }
            } else {
                float* Dp = (float*)Dst + (long)b * sD;
#pragma unroll
                for (int v = 0; v < 8; ++v) {
                    float x = acc[i][j][v] * scale + bb;
                    if (flags & 2) x = fmaxf(x, 0.f);
                    Dp[(long)(row0 + v) * Nn + col] = x;
                }
            }
        }
    }
}

__global__ void cvt_f32_bf16(const float* __restrict__ s, __bf16* __restrict__ d, long n)
{
    long i = (long)blockIdx.x * blockDim.x + threadIdx.x;
    if (i < n) d[i] = (__bf16)s[i];
}

// dst[b][c][r] = (bf16) src[b][r][c]   (src: batch x R x C)
__global__ void tr_f32_bf16(const float* __restrict__ s, __bf16* __restrict__ d,
                            int R, int C, int batch)
{
    long per = (long)R * C;
    long i = (long)blockIdx.x * blockDim.x + threadIdx.x;
    if (i >= per * batch) return;
    long b  = i / per;
    long j  = i - b * per;
    int  rr = (int)(j / C);
    int  cc = (int)(j - (long)rr * C);
    d[b * per + (long)cc * R + rr] = (__bf16)s[i];
}

// One block per score row: masked softmax over N=1024 columns, bf16 output.
__global__ void __launch_bounds__(256)
softmax_mask(const float* __restrict__ scores, const float* __restrict__ edges,
             __bf16* __restrict__ attn)
{
    const int NCOL = 1024;
    long row = blockIdx.x;
    const float* s = scores + row * NCOL;
    const float* e = edges  + row * NCOL;
    __bf16* a = attn + row * NCOL;
    __shared__ float red[256];
    int t = threadIdx.x;
    float v[4], msk[4];
    float mx = -3.4e38f;
#pragma unroll
    for (int j = 0; j < 4; ++j) {
        int idx = t + j * 256;
        float sv = s[idx];
        float ev = e[idx];
        msk[j] = ev;
        if (ev == 0.f) sv = -1e18f;     // mask = (rel_edges == 0)
        v[j] = sv;
        mx = fmaxf(mx, sv);
    }
    red[t] = mx; __syncthreads();
    for (int st = 128; st > 0; st >>= 1) {
        if (t < st) red[t] = fmaxf(red[t], red[t + st]);
        __syncthreads();
    }
    mx = red[0]; __syncthreads();
    float sum = 0.f;
#pragma unroll
    for (int j = 0; j < 4; ++j) { v[j] = expf(v[j] - mx); sum += v[j]; }
    red[t] = sum; __syncthreads();
    for (int st = 128; st > 0; st >>= 1) {
        if (t < st) red[t] += red[t + st];
        __syncthreads();
    }
    float inv = 1.f / red[0];
#pragma unroll
    for (int j = 0; j < 4; ++j) {
        int idx = t + j * 256;
        float av = (msk[j] == 0.f) ? 0.f : v[j] * inv;
        a[idx] = (__bf16)av;
    }
}

// One block per row: x = LayerNorm(x + ctx) * g + b, D=1024
__global__ void __launch_bounds__(256)
add_layernorm(float* __restrict__ x, const float* __restrict__ ctx,
              const float* __restrict__ g, const float* __restrict__ bta)
{
    const int Dd = 1024;
    long row = blockIdx.x;
    float* xr = x + row * Dd;
    const float* cr = ctx + row * Dd;
    __shared__ float red[256];
    int t = threadIdx.x;
    float u[4];
    float s = 0.f;
#pragma unroll
    for (int j = 0; j < 4; ++j) {
        int idx = t + j * 256;
        u[j] = xr[idx] + cr[idx];
        s += u[j];
    }
    red[t] = s; __syncthreads();
    for (int st = 128; st > 0; st >>= 1) { if (t < st) red[t] += red[t + st]; __syncthreads(); }
    float mu = red[0] * (1.f / 1024.f); __syncthreads();
    float vs = 0.f;
#pragma unroll
    for (int j = 0; j < 4; ++j) { float d = u[j] - mu; vs += d * d; }
    red[t] = vs; __syncthreads();
    for (int st = 128; st > 0; st >>= 1) { if (t < st) red[t] += red[t + st]; __syncthreads(); }
    float rinv = rsqrtf(red[0] * (1.f / 1024.f) + 1e-5f);
#pragma unroll
    for (int j = 0; j < 4; ++j) {
        int idx = t + j * 256;
        xr[idx] = (u[j] - mu) * rinv * g[idx] + bta[idx];
    }
}

static void gemm(hipStream_t st, const __bf16* A, const __bf16* BT, const float* bias,
                 void* Dst, int M, int Nn, int K, long sA, long sB, long sD,
                 int batch, float scale, int flags)
{
    long waves = (long)batch * (M >> 5) * (Nn >> 6);   // 32x64 block per wave
    int blocks = (int)((waves + WAVES_PER_BLOCK - 1) / WAVES_PER_BLOCK);
    gcn_wmma_gemm<<<blocks, 256, 0, st>>>(A, BT, bias, Dst, M, Nn, K,
                                          sA, sB, sD, batch, scale, flags);
}

extern "C" void kernel_launch(void* const* d_in, const int* /*in_sizes*/, int /*n_in*/,
                              void* d_out, int /*out_size*/, void* d_ws, size_t /*ws_size*/,
                              hipStream_t stream)
{
    const int B = 32, N = 1024, DIN = 1024, D = 1024, F = 512, L = 2;
    const int M = B * N;

    const float* node  = (const float*)d_in[0];
    const float* edges = (const float*)d_in[1];
    const float* Wemb  = (const float*)d_in[2];
    const float* bemb  = (const float*)d_in[3];
    const float* Wq    = (const float*)d_in[4];
    const float* bq    = (const float*)d_in[5];
    const float* Wk    = (const float*)d_in[6];
    const float* bk    = (const float*)d_in[7];
    const float* Wc    = (const float*)d_in[8];
    const float* lng   = (const float*)d_in[9];
    const float* lnb   = (const float*)d_in[10];
    float* x = (float*)d_out;   // x lives in d_out throughout

    char* w = (char*)d_ws;
    __bf16* x_bf  = (__bf16*)w;  w += (size_t)M * D * 2;       // x, bf16 row-major
    __bf16* xT_bf = (__bf16*)w;  w += (size_t)M * D * 2;       // x^T per batch (D x N)
    __bf16* q_bf  = (__bf16*)w;  w += (size_t)M * F * 2;
    __bf16* k_bf  = (__bf16*)w;  w += (size_t)M * F * 2;
    float*  sc    = (float*)w;   w += (size_t)B * N * N * 4;   // scores, reused as ctx2 f32
    __bf16* at_bf = (__bf16*)w;  w += (size_t)B * N * N * 2;
    __bf16* cx_bf = (__bf16*)w;  w += (size_t)M * D * 2;
    __bf16* WembT = (__bf16*)w;  w += (size_t)DIN * D * 2;
    __bf16* WqT   = (__bf16*)w;  w += (size_t)L * D * F * 2;
    __bf16* WkT   = (__bf16*)w;  w += (size_t)L * D * F * 2;
    __bf16* WcT   = (__bf16*)w;  w += (size_t)L * D * D * 2;

    // ---- pack weights (transpose + f32->bf16) ----
    { long n = (long)DIN * D; tr_f32_bf16<<<(n + 255) / 256, 256, 0, stream>>>(Wemb, WembT, DIN, D, 1); }
    { long n = (long)L * D * F; tr_f32_bf16<<<(n + 255) / 256, 256, 0, stream>>>(Wq, WqT, D, F, L); }
    { long n = (long)L * D * F; tr_f32_bf16<<<(n + 255) / 256, 256, 0, stream>>>(Wk, WkT, D, F, L); }
    { long n = (long)L * D * D; tr_f32_bf16<<<(n + 255) / 256, 256, 0, stream>>>(Wc, WcT, D, D, L); }

    // ---- embedding: x = relu(node @ Wemb + bemb) ----
    { long n = (long)M * DIN; cvt_f32_bf16<<<(n + 255) / 256, 256, 0, stream>>>(node, x_bf, n); }
    gemm(stream, x_bf, WembT, bemb, x, M, D, DIN, 0, 0, 0, 1, 1.f, 1 | 2);
    { long n = (long)M * D; cvt_f32_bf16<<<(n + 255) / 256, 256, 0, stream>>>(x, x_bf, n); }
    { long n = (long)M * D; tr_f32_bf16<<<(n + 255) / 256, 256, 0, stream>>>(x, xT_bf, N, D, B); }

    const float inv_sqrt_f = 0.044194173824159216f;  // 1/sqrt(512)

    for (int l = 0; l < L; ++l) {
        // q, k projections (bias, bf16 out)
        gemm(stream, x_bf, WqT + (size_t)l * D * F, bq + (size_t)l * F, q_bf,
             M, F, D, 0, 0, 0, 1, 1.f, 1 | 4);
        gemm(stream, x_bf, WkT + (size_t)l * D * F, bk + (size_t)l * F, k_bf,
             M, F, D, 0, 0, 0, 1, 1.f, 1 | 4);
        // scores = q @ k^T / sqrt(F)   (batched; k rows already give column slices)
        gemm(stream, q_bf, k_bf, nullptr, sc, N, N, F,
             (long)N * F, (long)N * F, (long)N * N, B, inv_sqrt_f, 0);
        // masked softmax -> bf16 attention
        softmax_mask<<<B * N, 256, 0, stream>>>(sc, edges, at_bf);
        // ctx = attn @ x   (batched; B operand from x^T) -> bf16
        gemm(stream, at_bf, xT_bf, nullptr, cx_bf, N, D, N,
             (long)N * N, (long)D * N, (long)N * D, B, 1.f, 4);
        // ctx2 = ctx @ Wc  (f32, into the scores buffer)
        gemm(stream, cx_bf, WcT + (size_t)l * D * D, nullptr, sc,
             M, D, D, 0, 0, 0, 1, 1.f, 0);
        // x = LN(x + ctx2) * g + b
        add_layernorm<<<M, 256, 0, stream>>>(x, sc, lng + (size_t)l * D, lnb + (size_t)l * D);
        if (l + 1 < L) {
            long n = (long)M * D;
            cvt_f32_bf16<<<(n + 255) / 256, 256, 0, stream>>>(x, x_bf, n);
            tr_f32_bf16<<<(n + 255) / 256, 256, 0, stream>>>(x, xT_bf, N, D, B);
        }
    }
}
